// MLA_64931315581702
// MI455X (gfx1250) — matrix operand order repaired and checked by reading, not compiled, not run on hardware
//
#include <hip/hip_runtime.h>
#include <hip/hip_bf16.h>
#include <math.h>

// ---------------- problem constants ----------------
#define TT      2048
#define DD      4096
#define NHEAD   32
#define Q_LORA  1536
#define KV_LORA 512
#define QK_NOPE 128
#define QK_ROPE 64
#define QK_HEAD 192           // 128 + 64
#define V_HEAD  128
#define KV_DOWN_N 576         // 512 + 64
#define QUP_N   6144          // 32*192
#define KVUP_N  8192          // 32*256
#define EPSF    1e-6f

typedef __attribute__((ext_vector_type(16))) __bf16 v16bf;
typedef __attribute__((ext_vector_type(8)))  float  v8f;

// A-matrix (16x32, 16-bit) element->K map used here for wave32 WMMA (A and B share it; not the ISA table order).
// e=0..15 -> k = 16*half + e          (contiguous, matches B fill)
__device__ __forceinline__ int a_kmap(int e, int half) {
    return (half << 4) + e;
}

#define WMMA_BF16(A, B, C) \
    __builtin_amdgcn_wmma_f32_16x16x32_bf16(false, (A), false, (B), (short)0, (C), false, false)

// ---- CDNA5 async global->LDS copy (16B per lane), tracked by ASYNCcnt ----
__device__ __forceinline__ void async_ld16(void* lds, const void* g) {
    asm volatile("global_load_async_to_lds_b128 %0, %1, off"
                 :: "v"((unsigned)(size_t)lds),
                    "v"((unsigned long long)(size_t)g)
                 : "memory");
}
__device__ __forceinline__ void wait_async0() {
    asm volatile("s_wait_asynccnt 0" ::: "memory");
}

// ======================================================================
// f32 -> bf16 flat convert, 4 elems/thread
// ======================================================================
__global__ __launch_bounds__(256)
void convert_bf16(const float* __restrict__ in, __bf16* __restrict__ out, int n4) {
    int i = blockIdx.x * 256 + threadIdx.x;
    if (i < n4) {
        float4 v = ((const float4*)in)[i];
        size_t b = (size_t)i * 4;
        out[b + 0] = (__bf16)v.x;
        out[b + 1] = (__bf16)v.y;
        out[b + 2] = (__bf16)v.z;
        out[b + 3] = (__bf16)v.w;
    }
}

// ======================================================================
// f32 [K,N] -> bf16 [N,K] tiled transpose-convert (weights)
// grid = (N/32, K/32), block = 256 (32x8)
// ======================================================================
__global__ __launch_bounds__(256)
void transpose_convert(const float* __restrict__ in, __bf16* __restrict__ out,
                       int K, int N) {
    __shared__ float tile[32][33];
    const int kb = blockIdx.y << 5, nb = blockIdx.x << 5;
    const int tx = threadIdx.x & 31, ty = threadIdx.x >> 5;
    for (int i = ty; i < 32; i += 8)
        tile[i][tx] = in[(size_t)(kb + i) * N + nb + tx];
    __syncthreads();
    for (int i = ty; i < 32; i += 8)
        out[(size_t)(nb + i) * K + kb + tx] = (__bf16)tile[tx][i];
}

// ======================================================================
// bf16 [T, 8192] (v part per head) -> bf16 vT[head][128][T]
// grid = (T/32, 128/32, NHEAD), block = 256 (32x8)
// ======================================================================
__global__ __launch_bounds__(256)
void transpose_v(const __bf16* __restrict__ kvu, __bf16* __restrict__ vt) {
    __shared__ __bf16 tile[32][33];
    const int head = blockIdx.z;
    const int sb = blockIdx.x << 5, hb = blockIdx.y << 5;
    const int tx = threadIdx.x & 31, ty = threadIdx.x >> 5;
    for (int i = ty; i < 32; i += 8)
        tile[i][tx] = kvu[(size_t)(sb + i) * KVUP_N + head * 256 + 128 + hb + tx];
    __syncthreads();
    for (int i = ty; i < 32; i += 8)
        vt[((size_t)head * V_HEAD + hb + i) * TT + sb + tx] = tile[tx][i];
}

// ======================================================================
// GEMM:  C[M,N] = A[M,K] * BT[N,K]^T   (bf16 in, OutT out, WMMA bf16)
// Block tile 64x64, 4 waves (each 32x32 = 2x2 frags), K-chunk 64.
// Double-buffered LDS, tiles fetched with async global->LDS copies.
// ======================================================================
template <typename OutT>
__global__ __launch_bounds__(128)
void gemm_bf16(const __bf16* __restrict__ A, const __bf16* __restrict__ BT,
               OutT* __restrict__ C, int M, int N, int K) {
    __shared__ __attribute__((aligned(16))) __bf16 As[2][64][72];
    __shared__ __attribute__((aligned(16))) __bf16 BsT[2][64][72];

    const int tid  = threadIdx.x;
    const int lane = tid & 31;
    const int wave = tid >> 5;
    const int half = lane >> 4;
    const int ln   = lane & 15;
    const int wm   = (wave >> 1) << 5;
    const int wn   = (wave & 1) << 5;
    const int m0   = blockIdx.y << 6;
    const int n0   = blockIdx.x << 6;

    v8f acc[2][2] = {};

    // stage tile 0 into buffer 0
    for (int i = tid; i < 64 * 8; i += 128) {
        int r = i >> 3, c8 = (i & 7) << 3;
        async_ld16(&As[0][r][c8],  A  + (size_t)(m0 + r) * K + c8);
        async_ld16(&BsT[0][r][c8], BT + (size_t)(n0 + r) * K + c8);
    }

    int buf = 0;
    for (int k0 = 0; k0 < K; k0 += 64) {
        wait_async0();          // this tile's async copies (own wave) done
        __syncthreads();        // all waves done; prev compute done
        if (k0 + 64 < K) {      // overlap: fetch next tile into other buffer
            for (int i = tid; i < 64 * 8; i += 128) {
                int r = i >> 3, c8 = (i & 7) << 3;
                async_ld16(&As[buf ^ 1][r][c8],
                           A + (size_t)(m0 + r) * K + k0 + 64 + c8);
                async_ld16(&BsT[buf ^ 1][r][c8],
                           BT + (size_t)(n0 + r) * K + k0 + 64 + c8);
            }
        }

#pragma unroll
        for (int kk = 0; kk < 64; kk += 32) {
            v16bf af[2], bfr[2];
#pragma unroll
            for (int e = 0; e < 16; ++e) {
                int km = kk + a_kmap(e, half);
                af[0][e]  = As[buf][wm + ln][km];
                af[1][e]  = As[buf][wm + 16 + ln][km];
                bfr[0][e] = BsT[buf][wn + ln][kk + (half << 4) + e];
                bfr[1][e] = BsT[buf][wn + 16 + ln][kk + (half << 4) + e];
            }
#pragma unroll
            for (int i = 0; i < 2; ++i)
#pragma unroll
                for (int j = 0; j < 2; ++j)
                    acc[i][j] = WMMA_BF16(af[i], bfr[j], acc[i][j]);
        }
        buf ^= 1;
    }

#pragma unroll
    for (int i = 0; i < 2; ++i)
#pragma unroll
        for (int j = 0; j < 2; ++j)
#pragma unroll
            for (int v = 0; v < 8; ++v) {
                int row = m0 + wm + (i << 4) + v + (half << 3);
                int col = n0 + wn + (j << 4) + ln;
                C[(size_t)row * N + col] = (OutT)acc[i][j][v];
            }
}

// ======================================================================
// Yarn rotary table: cos/sin[T][32]
// ======================================================================
__global__ __launch_bounds__(32)
void yarn_table(const int* __restrict__ positions,
                float* __restrict__ cosT, float* __restrict__ sinT) {
    const int t = blockIdx.x;
    const int i = threadIdx.x;      // 0..31
    const float TWO_PI = 6.28318530717958647692f;

    float pf        = powf(10000.f, (float)i / 32.f);
    float inv_extra = 1.f / pf;
    float inv_inter = 1.f / (40.f * pf);

    float denom = 2.f * logf(10000.f);
    float cl = 64.f * logf(4096.f / (32.f * TWO_PI)) / denom;
    float ch = 64.f * logf(4096.f / ( 1.f * TWO_PI)) / denom;
    float low  = fmaxf(floorf(cl), 0.f);
    float high = fminf(ceilf(ch), 63.f);
    float ramp = ((float)i - low) / fmaxf(high - low, 0.001f);
    ramp = fminf(fmaxf(ramp, 0.f), 1.f);
    float mask = 1.f - ramp;
    float inv_freq = inv_inter * (1.f - mask) + inv_extra * mask;
    float f = (float)positions[t] * inv_freq;
    cosT[t * 32 + i] = cosf(f);
    sinT[t * 32 + i] = sinf(f);
}

// ======================================================================
// RMS norm rows of x[rows, cols] (f32 in) -> bf16 out
// ======================================================================
__global__ __launch_bounds__(256)
void rmsnorm_to_bf16(const float* __restrict__ x, const float* __restrict__ scale,
                     __bf16* __restrict__ y, int cols) {
    __shared__ float red[256];
    const int row = blockIdx.x, tid = threadIdx.x;
    float ss = 0.f;
    for (int c = tid; c < cols; c += 256) {
        float v = x[(size_t)row * cols + c];
        ss += v * v;
    }
    red[tid] = ss;
    __syncthreads();
    for (int s = 128; s > 0; s >>= 1) {
        if (tid < s) red[tid] += red[tid + s];
        __syncthreads();
    }
    float inv = rsqrtf(red[0] / (float)cols + EPSF);
    for (int c = tid; c < cols; c += 256)
        y[(size_t)row * cols + c] = (__bf16)(x[(size_t)row * cols + c] * inv * scale[c]);
}

// ======================================================================
// kv path: rms-norm first 512 cols of kv_raw[T,576] -> kv_lat bf16[T,512];
// rope last 64 cols -> krope bf16[T,64]
// ======================================================================
__global__ __launch_bounds__(256)
void kv_norm_rope(const float* __restrict__ kvraw, const float* __restrict__ scale,
                  const float* __restrict__ cosT, const float* __restrict__ sinT,
                  __bf16* __restrict__ kvlat, __bf16* __restrict__ krope) {
    __shared__ float red[256];
    const int row = blockIdx.x, tid = threadIdx.x;
    float ss = 0.f;
    for (int c = tid; c < KV_LORA; c += 256) {
        float v = kvraw[(size_t)row * KV_DOWN_N + c];
        ss += v * v;
    }
    red[tid] = ss;
    __syncthreads();
    for (int s = 128; s > 0; s >>= 1) {
        if (tid < s) red[tid] += red[tid + s];
        __syncthreads();
    }
    float inv = rsqrtf(red[0] / (float)KV_LORA + EPSF);
    for (int c = tid; c < KV_LORA; c += 256)
        kvlat[(size_t)row * KV_LORA + c] =
            (__bf16)(kvraw[(size_t)row * KV_DOWN_N + c] * inv * scale[c]);

    if (tid < 32) {
        float x1 = kvraw[(size_t)row * KV_DOWN_N + KV_LORA + tid];
        float x2 = kvraw[(size_t)row * KV_DOWN_N + KV_LORA + 32 + tid];
        float c = cosT[row * 32 + tid], s = sinT[row * 32 + tid];
        krope[row * 64 + tid]      = (__bf16)(x1 * c - x2 * s);
        krope[row * 64 + 32 + tid] = (__bf16)(x1 * s + x2 * c);
    }
}

// ======================================================================
// In-place rope on q_full bf16[t, h*192 + 128 .. +192)
// ======================================================================
__global__ __launch_bounds__(32)
void q_rope_bf16(__bf16* __restrict__ qf, const float* __restrict__ cosT,
                 const float* __restrict__ sinT) {
    const int t = blockIdx.x, h = blockIdx.y, i = threadIdx.x; // i<32
    size_t base = (size_t)t * QUP_N + h * QK_HEAD + QK_NOPE;
    float x1 = (float)qf[base + i], x2 = (float)qf[base + 32 + i];
    float c = cosT[t * 32 + i], s = sinT[t * 32 + i];
    qf[base + i]      = (__bf16)(x1 * c - x2 * s);
    qf[base + 32 + i] = (__bf16)(x1 * s + x2 * c);
}

// ======================================================================
// Causal flash attention, all-bf16 operands, WMMA math.
// grid = (T/64, NHEAD), block = 128 (4 waves). Wave w owns query rows
// t0+16w..+15; 4 waves share K/V LDS tiles (32 keys / iter) staged with
// async global->LDS copies.
// ======================================================================
__global__ __launch_bounds__(128)
void mla_attn(const __bf16* __restrict__ qf,   // [T, 6144] rope applied
              const __bf16* __restrict__ kvu,  // [T, 8192] (k_nope|v per head)
              const __bf16* __restrict__ kr,   // [T, 64]
              const __bf16* __restrict__ vt,   // [NHEAD][128][T]
              __bf16* __restrict__ out,        // [T, 4096] (t, n*128+h)
              float sm_scale) {
    __shared__ __attribute__((aligned(16))) __bf16 Qs[64][200];   // q rows x 192
    __shared__ __attribute__((aligned(16))) __bf16 Ks[32][200];   // keys x 192
    __shared__ __attribute__((aligned(16))) __bf16 VsT[128][40];  // h x 32 keys
    __shared__ __attribute__((aligned(16))) __bf16 Ps[4][16][40]; // per-wave probs

    const int head = blockIdx.y;
    const int t0   = blockIdx.x << 6;
    const int tid  = threadIdx.x;
    const int wave = tid >> 5;
    const int lane = tid & 31;
    const int half = lane >> 4;
    const int ln   = lane & 15;
    const int tw   = t0 + (wave << 4);

    // Q tile: 64 x 192 halfs, async staged (covered by first wait in loop)
    for (int i = tid; i < 64 * 24; i += 128) {
        int r = i / 24, c8 = (i % 24) << 3;
        async_ld16(&Qs[r][c8],
                   qf + (size_t)(t0 + r) * QUP_N + head * QK_HEAD + c8);
    }

    float m_st[8], l_st[8];
    v8f O[8] = {};
#pragma unroll
    for (int v = 0; v < 8; ++v) { m_st[v] = -1e30f; l_st[v] = 0.f; }

    const int send = t0 + 64;
    for (int s0 = 0; s0 < send; s0 += 32) {
        __syncthreads();    // previous tile fully consumed
        // K tile: 32 x 192 (nope | rope), async; prefetch next s-tile to L2
        for (int i = tid; i < 32 * 24; i += 128) {
            int r = i / 24, c8 = (i % 24) << 3;
            const __bf16* src = (c8 < QK_NOPE)
                ? kvu + (size_t)(s0 + r) * KVUP_N + head * 256 + c8
                : kr + (size_t)(s0 + r) * 64 + (c8 - QK_NOPE);
            async_ld16(&Ks[r][c8], src);
            if (s0 + 32 < send)
                __builtin_prefetch(kvu + (size_t)(s0 + 32 + r) * KVUP_N +
                                   head * 256 + c8, 0, 0);
        }
        // V^T tile: 128 h x 32 keys, async
        for (int i = tid; i < 128 * 4; i += 128) {
            int h = i >> 2, c8 = (i & 3) << 3;
            async_ld16(&VsT[h][c8],
                       vt + ((size_t)head * V_HEAD + h) * TT + s0 + c8);
        }
        wait_async0();
        __syncthreads();

        if (s0 > tw + 15) continue;   // tile fully masked for this wave

        // ---- scores = Q K^T over 192 (two 16-col score tiles) ----
        v8f sc0 = {}, sc1 = {};
#pragma unroll
        for (int kk = 0; kk < 6; ++kk) {
            v16bf aq, bk0, bk1;
#pragma unroll
            for (int e = 0; e < 16; ++e) {
                aq[e]  = Qs[(wave << 4) + ln][kk * 32 + a_kmap(e, half)];
                bk0[e] = Ks[ln][kk * 32 + (half << 4) + e];
                bk1[e] = Ks[16 + ln][kk * 32 + (half << 4) + e];
            }
            sc0 = WMMA_BF16(aq, bk0, sc0);
            sc1 = WMMA_BF16(aq, bk1, sc1);
        }

        // ---- online softmax (row = v + 8*half, cols across 16 lanes) ----
#pragma unroll
        for (int v = 0; v < 8; ++v) {
            int t = tw + v + (half << 3);
            bool ok_a = (s0 + ln)      <= t;
            bool ok_b = (s0 + 16 + ln) <= t;
            float s_a = ok_a ? sc0[v] * sm_scale : -1e30f;
            float s_b = ok_b ? sc1[v] * sm_scale : -1e30f;
            float rmax = fmaxf(s_a, s_b);
#pragma unroll
            for (int off = 1; off < 16; off <<= 1)
                rmax = fmaxf(rmax, __shfl_xor(rmax, off, 32));
            float mn = fmaxf(m_st[v], rmax);
            float p_a = ok_a ? __expf(s_a - mn) : 0.f;
            float p_b = ok_b ? __expf(s_b - mn) : 0.f;
            float rs = p_a + p_b;
#pragma unroll
            for (int off = 1; off < 16; off <<= 1)
                rs += __shfl_xor(rs, off, 32);
            float alpha = __expf(m_st[v] - mn);
            l_st[v] = l_st[v] * alpha + rs;
            m_st[v] = mn;
#pragma unroll
            for (int nt = 0; nt < 8; ++nt) O[nt][v] *= alpha;
            Ps[wave][v + (half << 3)][ln]      = (__bf16)p_a;
            Ps[wave][v + (half << 3)][16 + ln] = (__bf16)p_b;
        }

        // ---- O += P V  (K-dim 32, eight 16-col output tiles) ----
        v16bf ap;
#pragma unroll
        for (int e = 0; e < 16; ++e)
            ap[e] = Ps[wave][ln][a_kmap(e, half)];
#pragma unroll
        for (int nt = 0; nt < 8; ++nt) {
            v16bf bv;
#pragma unroll
            for (int e = 0; e < 16; ++e)
                bv[e] = VsT[(nt << 4) + ln][(half << 4) + e];
            O[nt] = WMMA_BF16(ap, bv, O[nt]);
        }
    }

    // ---- epilogue: normalize, store bf16 [t, head*128 + h] ----
#pragma unroll
    for (int nt = 0; nt < 8; ++nt)
#pragma unroll
        for (int v = 0; v < 8; ++v) {
            float d = l_st[v];
            float val = (d > 0.f) ? O[nt][v] / d : 0.f;
            int row = tw + v + (half << 3);
            out[(size_t)row * (NHEAD * V_HEAD) + head * V_HEAD + (nt << 4) + ln] =
                (__bf16)val;
        }
}

// ======================================================================
// Launcher
// ======================================================================
extern "C" void kernel_launch(void* const* d_in, const int* in_sizes, int n_in,
                              void* d_out, int out_size, void* d_ws, size_t ws_size,
                              hipStream_t stream) {
    (void)in_sizes; (void)n_in; (void)out_size; (void)ws_size;

    const float* x             = (const float*)d_in[0];
    const int*   positions     = (const int*)  d_in[1];
    const float* w_q_down      = (const float*)d_in[2];
    const float* q_norm_scale  = (const float*)d_in[3];
    const float* w_q_up        = (const float*)d_in[4];
    const float* w_kv_down     = (const float*)d_in[5];
    const float* kv_norm_scale = (const float*)d_in[6];
    const float* w_kv_up       = (const float*)d_in[7];
    const float* w_o           = (const float*)d_in[8];
    float* out = (float*)d_out;

    char* ws = (char*)d_ws;
    float*  cosT    = (float*)ws;  ws += (size_t)TT * 32 * sizeof(float);
    float*  sinT    = (float*)ws;  ws += (size_t)TT * 32 * sizeof(float);
    __bf16* x_bf    = (__bf16*)ws; ws += (size_t)TT * DD * 2;
    __bf16* wqdT    = (__bf16*)ws; ws += (size_t)DD * Q_LORA * 2;
    __bf16* wkvdT   = (__bf16*)ws; ws += (size_t)DD * KV_DOWN_N * 2;
    __bf16* wquT    = (__bf16*)ws; ws += (size_t)Q_LORA * QUP_N * 2;
    __bf16* wkvuT   = (__bf16*)ws; ws += (size_t)KV_LORA * KVUP_N * 2;
    __bf16* woT     = (__bf16*)ws; ws += (size_t)DD * DD * 2;
    float*  q_raw   = (float*)ws;  ws += (size_t)TT * Q_LORA * sizeof(float);
    float*  kv_raw  = (float*)ws;  ws += (size_t)TT * KV_DOWN_N * sizeof(float);
    __bf16* q_lat   = (__bf16*)ws; ws += (size_t)TT * Q_LORA * 2;
    __bf16* kv_lat  = (__bf16*)ws; ws += (size_t)TT * KV_LORA * 2;
    __bf16* krope   = (__bf16*)ws; ws += (size_t)TT * 64 * 2;
    __bf16* q_full  = (__bf16*)ws; ws += (size_t)TT * QUP_N * 2;
    __bf16* kvu     = (__bf16*)ws; ws += (size_t)TT * KVUP_N * 2;
    __bf16* vT      = (__bf16*)ws; ws += (size_t)NHEAD * V_HEAD * TT * 2;
    __bf16* attn_bf = (__bf16*)ws; ws += (size_t)TT * NHEAD * V_HEAD * 2;

    // ---- precision / layout prep ----
    convert_bf16<<<(TT * DD / 4 + 255) / 256, 256, 0, stream>>>(x, x_bf, TT * DD / 4);
    transpose_convert<<<dim3(Q_LORA / 32, DD / 32), 256, 0, stream>>>(
        w_q_down, wqdT, DD, Q_LORA);
    transpose_convert<<<dim3(KV_DOWN_N / 32, DD / 32), 256, 0, stream>>>(
        w_kv_down, wkvdT, DD, KV_DOWN_N);
    transpose_convert<<<dim3(QUP_N / 32, Q_LORA / 32), 256, 0, stream>>>(
        w_q_up, wquT, Q_LORA, QUP_N);
    transpose_convert<<<dim3(KVUP_N / 32, KV_LORA / 32), 256, 0, stream>>>(
        w_kv_up, wkvuT, KV_LORA, KVUP_N);
    transpose_convert<<<dim3(DD / 32, DD / 32), 256, 0, stream>>>(
        w_o, woT, DD, DD);

    yarn_table<<<TT, 32, 0, stream>>>(positions, cosT, sinT);

    // ---- down-projections ----
    gemm_bf16<float><<<dim3(Q_LORA / 64, TT / 64), 128, 0, stream>>>(
        x_bf, wqdT, q_raw, TT, Q_LORA, DD);
    gemm_bf16<float><<<dim3(KV_DOWN_N / 64, TT / 64), 128, 0, stream>>>(
        x_bf, wkvdT, kv_raw, TT, KV_DOWN_N, DD);

    // ---- norms + k-rope ----
    rmsnorm_to_bf16<<<TT, 256, 0, stream>>>(q_raw, q_norm_scale, q_lat, Q_LORA);
    kv_norm_rope<<<TT, 256, 0, stream>>>(kv_raw, kv_norm_scale, cosT, sinT,
                                         kv_lat, krope);

    // ---- up-projections ----
    gemm_bf16<__bf16><<<dim3(QUP_N / 64, TT / 64), 128, 0, stream>>>(
        q_lat, wquT, q_full, TT, QUP_N, Q_LORA);
    q_rope_bf16<<<dim3(TT, NHEAD), 32, 0, stream>>>(q_full, cosT, sinT);
    gemm_bf16<__bf16><<<dim3(KVUP_N / 64, TT / 64), 128, 0, stream>>>(
        kv_lat, wkvuT, kvu, TT, KVUP_N, KV_LORA);
    transpose_v<<<dim3(TT / 32, V_HEAD / 32, NHEAD), 256, 0, stream>>>(kvu, vT);

    // ---- attention ----
    float yms = 0.1f * logf(40.f) + 1.f;   // yarn mscale(40, 1.0)
    float sm_scale = (1.f / sqrtf((float)QK_HEAD)) * yms * yms;
    mla_attn<<<dim3(TT / 64, NHEAD), 128, 0, stream>>>(
        q_full, kvu, krope, vT, attn_bf, sm_scale);

    // ---- output projection ----
    gemm_bf16<float><<<dim3(DD / 64, TT / 64), 128, 0, stream>>>(
        attn_bf, woT, out, TT, DD, DD);
}
